// GCN_45397804318999
// MI455X (gfx1250) — matrix operand order repaired and checked by reading, not compiled, not run on hardware
//
#include <hip/hip_runtime.h>

// ---------------------------------------------------------------------------
// GCN forward for MI455X (gfx1250, wave32, WMMA).
//   B=32, N=4096, F_IN=128, F_HID=64, F_LAT=32
//   Layer2 as one GEMM: H(4096 x 2048) = adj(4096x4096) @ S(4096x2048)
//   Split-bf16 (hi+lo) fp32 emulation: 3x v_wmma_f32_16x16x32_bf16 per tile.
//   adj/S pre-split into bf16 planes; hot GEMM loop does NO conversions,
//   double-buffered LDS staged with GLOBAL_LOAD_ASYNC_TO_LDS_B128 (ASYNCcnt)
//   when the toolchain exposes it, fragments are pure ds_load_b128.
// ---------------------------------------------------------------------------

typedef __bf16 bf16;
typedef __attribute__((ext_vector_type(8)))  __bf16 bf16x8;
typedef __attribute__((ext_vector_type(16))) __bf16 bf16x16;
typedef __attribute__((ext_vector_type(8)))  float  f32x8;
typedef __attribute__((ext_vector_type(4)))  int    i32x4;

#define GCN_B     32
#define GCN_N     4096
#define GCN_FIN   128
#define GCN_FHID  64
#define GCN_FLAT  32
#define GCN_COLS  (GCN_B * GCN_FHID)   // 2048
#define LDS_K     40                   // padded bf16 row stride (80B, 16B-aligned)

#define WMMA_BF16(A, Bm, C) \
  __builtin_amdgcn_wmma_f32_16x16x32_bf16(false, (A), false, (Bm), (short)0, (C), false, false)

// ---- Async global->LDS copy (CDNA5), with safe fallback --------------------
#if __has_builtin(__builtin_amdgcn_global_load_async_to_lds_b128)
#define GCN_ASYNC_LDS 1
__device__ __forceinline__ void cp16_async(const bf16* g, bf16* l) {
  // Builtin signature (from hipcc diagnostic): (v4i global*, v4i lds*, imm, imm)
  __builtin_amdgcn_global_load_async_to_lds_b128(
      (__attribute__((address_space(1))) i32x4*)(g),
      (__attribute__((address_space(3))) i32x4*)(l), 0, 0);
}
__device__ __forceinline__ void wait_async0() {
#if __has_builtin(__builtin_amdgcn_s_wait_asynccnt)
  __builtin_amdgcn_s_wait_asynccnt(0);
#else
  asm volatile("s_wait_asynccnt 0" ::: "memory");
#endif
}
#else
#define GCN_ASYNC_LDS 0
__device__ __forceinline__ void cp16_async(const bf16* g, bf16* l) {
  *(bf16x8*)l = *(const bf16x8*)g;   // sync fallback: reg round-trip
}
__device__ __forceinline__ void wait_async0() {}
#endif

// ---------------------------------------------------------------------------
// Fragment builders: contiguous 16B LDS loads in ISA-native WMMA layout.
// A (16x32): lanes 0-15 M=lane, K={0..7,16..23}; lanes 16-31 M=lane-16,
//            K={8..15,24..31}. Plane stored row-major [m][LDS_K].
// B (32x16): lanes 0-15 col=lane K={0..7,16..23}; lanes 16-31 col=lane-16 K={8..15,24..31}.
//            Plane stored K-contiguous [c][LDS_K].
// ---------------------------------------------------------------------------
__device__ __forceinline__ bf16x16 frag_a(const bf16* __restrict__ plane, int lane) {
  const int m  = lane & 15;
  const int kb = (lane & 16) ? 8 : 0;
  const bf16* r = plane + m * LDS_K + kb;
  bf16x8 p0 = *(const bf16x8*)(r);        // K = kb .. kb+7
  bf16x8 p1 = *(const bf16x8*)(r + 16);   // K = kb+16 .. kb+23
  return __builtin_shufflevector(p0, p1, 0,1,2,3,4,5,6,7,8,9,10,11,12,13,14,15);
}

__device__ __forceinline__ bf16x16 frag_b(const bf16* __restrict__ plane, int lane) {
  const int n  = lane & 15;
  const int kb = (lane & 16) ? 8 : 0;
  const bf16* r = plane + n * LDS_K + kb;
  bf16x8 p0 = *(const bf16x8*)(r);        // K = kb .. kb+7
  bf16x8 p1 = *(const bf16x8*)(r + 16);   // K = kb+16 .. kb+23
  return __builtin_shufflevector(p0, p1, 0,1,2,3,4,5,6,7,8,9,10,11,12,13,14,15);
}

// ---------------------------------------------------------------------------
// Kernel 0: split adj fp32 -> bf16 planes adjH/adjL (layout [m][k], like adj).
// ---------------------------------------------------------------------------
__global__ __launch_bounds__(256) void gcn_split_adj(const float* __restrict__ adj,
                                                     bf16* __restrict__ adjH,
                                                     bf16* __restrict__ adjL) {
  const size_t i = ((size_t)blockIdx.x * 256 + threadIdx.x) * 8;
  const float4 a = *(const float4*)(adj + i);
  const float4 b = *(const float4*)(adj + i + 4);
  float f[8] = {a.x, a.y, a.z, a.w, b.x, b.y, b.z, b.w};
  bf16x8 h, l;
#pragma unroll
  for (int v = 0; v < 8; ++v) {
    bf16 hv = (bf16)f[v];
    h[v] = hv;
    l[v] = (bf16)(f[v] - (float)hv);
  }
  *(bf16x8*)(adjH + i) = h;
  *(bf16x8*)(adjL + i) = l;
}

// ---------------------------------------------------------------------------
// Kernel 1: S = x @ W1 (split-bf16 WMMA, split done on the fly: only 4 k-steps)
// Output stored TRANSPOSED as bf16 planes: SHt/SLt[col][m], col = b*64+c.
// ---------------------------------------------------------------------------
__global__ __launch_bounds__(256) void gcn_gemm1(const float* __restrict__ x,
                                                 const float* __restrict__ W1,
                                                 bf16* __restrict__ SHt,
                                                 bf16* __restrict__ SLt) {
  __shared__ float lds_a[128 * 33];  // x tile [m][k]
  __shared__ float lds_b[32 * 65];   // W1 tile [k][c]

  const int tid  = threadIdx.x;
  const int lane = tid & 31;
  const int wave = tid >> 5;
  const int wm   = wave & 3;
  const int wc   = wave >> 2;
  const int r0   = blockIdx.x * 128;      // flat row = b*4096 + m
  const int b    = r0 >> 12;
  const int m0   = r0 & (GCN_N - 1);

  f32x8 acc[2][2] = {};

  for (int k0 = 0; k0 < GCN_FIN; k0 += 32) {
#pragma unroll
    for (int i = 0; i < 4; ++i) {
      int slot = tid + i * 256;
      int row  = slot >> 3;
      int kq   = (slot & 7) << 2;
      const float4 v = *(const float4*)(x + (size_t)(r0 + row) * GCN_FIN + k0 + kq);
      float* dst = lds_a + row * 33 + kq;
      dst[0] = v.x; dst[1] = v.y; dst[2] = v.z; dst[3] = v.w;
    }
#pragma unroll
    for (int i = 0; i < 2; ++i) {
      int slot = tid + i * 256;
      int row  = slot >> 4;
      int cq   = (slot & 15) << 2;
      const float4 v = *(const float4*)(W1 + (size_t)(k0 + row) * GCN_FHID + cq);
      float* dst = lds_b + row * 65 + cq;
      dst[0] = v.x; dst[1] = v.y; dst[2] = v.z; dst[3] = v.w;
    }
    __syncthreads();

    bf16x16 ah[2], al[2], bh[2], bl[2];
#pragma unroll
    for (int mi = 0; mi < 2; ++mi) {
      const int m  = lane & 15;
      const int kb = (lane & 16) ? 8 : 0;
      const float* row = lds_a + (wm * 32 + mi * 16 + m) * 33;
#pragma unroll
      for (int j = 0; j < 8; ++j) {
        float f0 = row[kb + j];
        float f1 = row[kb + 16 + j];
        bf16 h0 = (bf16)f0, h1 = (bf16)f1;
        ah[mi][j] = h0;            al[mi][j] = (bf16)(f0 - (float)h0);
        ah[mi][8 + j] = h1;        al[mi][8 + j] = (bf16)(f1 - (float)h1);
      }
    }
#pragma unroll
    for (int ci = 0; ci < 2; ++ci) {
      const int n  = lane & 15;
      const int kb = (lane & 16) ? 8 : 0;
      const float* col = lds_b + (wc * 32 + ci * 16 + n);
#pragma unroll
      for (int j = 0; j < 16; ++j) {
        float f = col[(kb + j + (j & 8)) * 65];
        bf16 h = (bf16)f;
        bh[ci][j] = h;
        bl[ci][j] = (bf16)(f - (float)h);
      }
    }
#pragma unroll
    for (int mi = 0; mi < 2; ++mi)
#pragma unroll
      for (int ci = 0; ci < 2; ++ci) {
        acc[mi][ci] = WMMA_BF16(ah[mi], bh[ci], acc[mi][ci]);
        acc[mi][ci] = WMMA_BF16(al[mi], bh[ci], acc[mi][ci]);
        acc[mi][ci] = WMMA_BF16(ah[mi], bl[ci], acc[mi][ci]);
      }
    __syncthreads();
  }

  const int n     = lane & 15;
  const int rhalf = (lane & 16) ? 8 : 0;
#pragma unroll
  for (int mi = 0; mi < 2; ++mi)
#pragma unroll
    for (int ci = 0; ci < 2; ++ci) {
      const int c      = wc * 32 + ci * 16 + n;
      const size_t col = (size_t)(b * GCN_FHID + c);
      const int m_base = m0 + wm * 32 + mi * 16 + rhalf;   // multiple of 8
      bf16x8 hv, lv;
#pragma unroll
      for (int v = 0; v < 8; ++v) {
        float f = acc[mi][ci][v];
        bf16 h = (bf16)f;
        hv[v] = h;
        lv[v] = (bf16)(f - (float)h);
      }
      *(bf16x8*)(SHt + col * GCN_N + m_base) = hv;
      *(bf16x8*)(SLt + col * GCN_N + m_base) = lv;
    }
}

// ---------------------------------------------------------------------------
// Kernel 2 (hot): H = relu(adj @ S + b1).
// Double-buffered LDS; tile k+1 staged asynchronously while computing tile k.
// Grid: (32 mtiles, 32 ctiles) x 256 threads. Block tile 128x64, K=4096.
// ---------------------------------------------------------------------------
__global__ __launch_bounds__(256) void gcn_gemm2(const bf16* __restrict__ adjH,
                                                 const bf16* __restrict__ adjL,
                                                 const bf16* __restrict__ SHt,
                                                 const bf16* __restrict__ SLt,
                                                 const float* __restrict__ b1,
                                                 float* __restrict__ H) {
  __shared__ bf16 ldsAH[2][128 * LDS_K];   // adj hi tile [m][k], ping-pong
  __shared__ bf16 ldsAL[2][128 * LDS_K];   // adj lo tile
  __shared__ bf16 ldsBH[2][64 * LDS_K];    // S hi tile [c][k]
  __shared__ bf16 ldsBL[2][64 * LDS_K];    // S lo tile

  const int tid  = threadIdx.x;
  const int lane = tid & 31;
  const int wave = tid >> 5;
  const int wm   = wave & 3;
  const int wc   = wave >> 2;
  const int m0   = blockIdx.x * 128;
  const int col0 = blockIdx.y * 64;        // == b*64

  // Per-thread staging slots (16B each).
  const int arow0 = (tid + 0)   >> 2, akq0 = ((tid + 0)   & 3) << 3;
  const int arow1 = (tid + 256) >> 2, akq1 = ((tid + 256) & 3) << 3;
  const int bcol  = tid >> 2,         bkq  = (tid & 3) << 3;

  #define STAGE_TILES(kk, pp)                                                  \
    do {                                                                       \
      size_t ga0 = (size_t)(m0 + arow0) * GCN_N + (kk) + akq0;                 \
      size_t ga1 = (size_t)(m0 + arow1) * GCN_N + (kk) + akq1;                 \
      size_t gb  = (size_t)(col0 + bcol) * GCN_N + (kk) + bkq;                 \
      cp16_async(adjH + ga0, &ldsAH[pp][arow0 * LDS_K + akq0]);                \
      cp16_async(adjL + ga0, &ldsAL[pp][arow0 * LDS_K + akq0]);                \
      cp16_async(adjH + ga1, &ldsAH[pp][arow1 * LDS_K + akq1]);                \
      cp16_async(adjL + ga1, &ldsAL[pp][arow1 * LDS_K + akq1]);                \
      cp16_async(SHt + gb,  &ldsBH[pp][bcol * LDS_K + bkq]);                   \
      cp16_async(SLt + gb,  &ldsBL[pp][bcol * LDS_K + bkq]);                   \
    } while (0)

  f32x8 acc[2][2] = {};

  // Prologue: stage tile 0 into buffer 0.
  STAGE_TILES(0, 0);
  wait_async0();
  __syncthreads();

#pragma unroll 2
  for (int k0 = 0; k0 < GCN_N; k0 += 32) {
    const int p = (k0 >> 5) & 1;

    // Stage next tile into the other buffer while this one computes.
    if (k0 + 32 < GCN_N) STAGE_TILES(k0 + 32, 1 - p);

    const bf16* aH = ldsAH[p];
    const bf16* aL = ldsAL[p];
    const bf16* bH = ldsBH[p];
    const bf16* bL = ldsBL[p];

    bf16x16 ah[2], al[2], bh[2], bl[2];
#pragma unroll
    for (int mi = 0; mi < 2; ++mi) {
      ah[mi] = frag_a(aH + (wm * 32 + mi * 16) * LDS_K, lane);
      al[mi] = frag_a(aL + (wm * 32 + mi * 16) * LDS_K, lane);
    }
#pragma unroll
    for (int ci = 0; ci < 2; ++ci) {
      bh[ci] = frag_b(bH + (wc * 32 + ci * 16) * LDS_K, lane);
      bl[ci] = frag_b(bL + (wc * 32 + ci * 16) * LDS_K, lane);
    }
#pragma unroll
    for (int mi = 0; mi < 2; ++mi)
#pragma unroll
      for (int ci = 0; ci < 2; ++ci) {
        acc[mi][ci] = WMMA_BF16(ah[mi], bh[ci], acc[mi][ci]);
        acc[mi][ci] = WMMA_BF16(al[mi], bh[ci], acc[mi][ci]);
        acc[mi][ci] = WMMA_BF16(ah[mi], bl[ci], acc[mi][ci]);
      }

    wait_async0();     // next tile's async copies landed in LDS
    __syncthreads();   // everyone done reading buf[p] / writing buf[1-p]
  }
  #undef STAGE_TILES

  // Epilogue: bias + relu, store H fp32 [n][col].
  const int n     = lane & 15;
  const int rhalf = (lane & 16) ? 8 : 0;
#pragma unroll
  for (int mi = 0; mi < 2; ++mi)
#pragma unroll
    for (int ci = 0; ci < 2; ++ci) {
      const int c = wc * 32 + ci * 16 + n;      // 0..63 -> b1 index
      const float bias = b1[c];
#pragma unroll
      for (int v = 0; v < 8; ++v) {
        const int m = m0 + wm * 32 + mi * 16 + rhalf + v;
        float hv = acc[mi][ci][v] + bias;
        hv = fmaxf(hv, 0.0f);
        H[(size_t)m * GCN_COLS + col0 + c] = hv;
      }
    }
}

// ---------------------------------------------------------------------------
// Kernel 3: agg[b,c] = sum_n adj[q_b, n] * H[n, b*64+c]; out = agg@W2 + b2.
// ---------------------------------------------------------------------------
__global__ __launch_bounds__(256) void gcn_final(const float* __restrict__ adj,
                                                 const int* __restrict__ q_ids,
                                                 const float* __restrict__ H,
                                                 const float* __restrict__ W2,
                                                 const float* __restrict__ b2,
                                                 float* __restrict__ out) {
  __shared__ float red[256];
  __shared__ float aggs[GCN_FHID];

  const int b   = blockIdx.x;
  const int tid = threadIdx.x;
  const int c   = tid & 63;
  const int g   = tid >> 6;
  const int qb  = q_ids[b];

  float acc = 0.0f;
  for (int nn = g; nn < GCN_N; nn += 4) {
    acc += adj[(size_t)qb * GCN_N + nn] * H[(size_t)nn * GCN_COLS + b * GCN_FHID + c];
  }
  red[tid] = acc;
  __syncthreads();
  if (g == 0) aggs[c] = red[c] + red[64 + c] + red[128 + c] + red[192 + c];
  __syncthreads();

  if (tid < GCN_FLAT) {
    float o = b2[tid];
#pragma unroll 8
    for (int cc = 0; cc < GCN_FHID; ++cc) o += aggs[cc] * W2[cc * GCN_FLAT + tid];
    out[b * GCN_FLAT + tid] = o;
  }
}

// ---------------------------------------------------------------------------
extern "C" void kernel_launch(void* const* d_in, const int* in_sizes, int n_in,
                              void* d_out, int out_size, void* d_ws, size_t ws_size,
                              hipStream_t stream) {
  const float* x     = (const float*)d_in[0];
  const int*   q_ids = (const int*)d_in[1];
  const float* adj   = (const float*)d_in[2];
  const float* W1    = (const float*)d_in[3];
  const float* b1    = (const float*)d_in[4];
  const float* W2    = (const float*)d_in[5];
  const float* b2    = (const float*)d_in[6];
  float* out = (float*)d_out;

  // Workspace layout (128 MB total):
  //   adjH 32MB | adjL 32MB | SHt 16MB | SLt 16MB | H 32MB
  char* ws = (char*)d_ws;
  bf16* adjH = (bf16*)(ws);
  bf16* adjL = (bf16*)(ws + (size_t)GCN_N * GCN_N * 2);
  bf16* SHt  = (bf16*)(ws + (size_t)GCN_N * GCN_N * 4);
  bf16* SLt  = (bf16*)(ws + (size_t)GCN_N * GCN_N * 4 + (size_t)GCN_COLS * GCN_N * 2);
  float* H   = (float*)(ws + (size_t)GCN_N * GCN_N * 4 + (size_t)GCN_COLS * GCN_N * 4);

  gcn_split_adj<<<dim3((GCN_N * (size_t)GCN_N) / (256 * 8)), 256, 0, stream>>>(adj, adjH, adjL);
  gcn_gemm1<<<dim3((GCN_B * GCN_N) / 128), 256, 0, stream>>>(x, W1, SHt, SLt);
  gcn_gemm2<<<dim3(GCN_N / 128, GCN_COLS / 64), 256, 0, stream>>>(adjH, adjL, SHt, SLt, b1, H);
  gcn_final<<<dim3(GCN_B), 256, 0, stream>>>(adj, q_ids, H, W2, b2, out);
}
